// Conv1_21328807592391
// MI455X (gfx1250) — matrix operand order repaired and checked by reading, not compiled, not run on hardware
//
#include <hip/hip_runtime.h>
#include <hip/hip_bf16.h>
#include <math.h>

// ---------------------------------------------------------------------------
// MI455X / gfx1250 implementation.
//
// B=2, C=64, N=4096, K=16 (+self loop -> KN=17).
// ~44 GFLOP of 64x128 GEMMs over 139k columns; <6 MB of real I/O.
// => compute bound: all matmuls via v_wmma_f32_16x16x32_f16 (wave32),
//    64x16 output tile per wave, W fragments register-resident,
//    all LDS operands channel-contiguous so B-fragments are ds_load_b128.
//
// Phases (ordered on `stream`):
//   k_init : zero 10 score accumulators
//   k1     : A[c,n,k] = sum_b relu(W_nn @ [x_i;x_j])  (alpha logits, 17.8MB ws)
//   k2     : 6 pair + 10 triple convs per batch fused with
//            softmax(leakyrelu(att_t*A)) dot -> atomicAdd S[10]
//   k3     : argmax over S[10] -> idx
//   k4     : recompute winning triple, W_out conv, relu, max over K -> d_out
// ---------------------------------------------------------------------------

typedef __attribute__((ext_vector_type(16))) _Float16 v16h;
typedef __attribute__((ext_vector_type(8)))  _Float16 v8h;
typedef __attribute__((ext_vector_type(8)))  float    v8f;

#define CH      64      // channels
#define CH2     128     // 2*CH (W_nn inner dim)
#define KN      17      // neighbors incl self loop
#define KE      16      // raw edge K
#define NB      16      // nodes per block
#define COLS    272     // NB*KN columns per block
#define NTILE   17      // COLS/16 WMMA column tiles
#define NNODES  4096
#define NBATCH  2
#define NTHREADS 256

// triple (i,j,kk) enumeration order matching the reference loops
__constant__ int c_TI[10] = {0,0,0,0,0,0,1,1,1,2};
__constant__ int c_TJ[10] = {1,1,1,2,2,3,2,2,3,3};
__constant__ int c_TK[10] = {2,3,4,3,4,4,3,4,4,4};

// ---------------- vector helpers -------------------------------------------
__device__ __forceinline__ v16h ld16(const _Float16* p) {   // 2x ds_load_b128
  v8h lo = *(const v8h*)p;
  v8h hi = *(const v8h*)(p + 8);
  return __builtin_shufflevector(lo, hi, 0,1,2,3,4,5,6,7,8,9,10,11,12,13,14,15);
}
__device__ __forceinline__ v16h splat16(_Float16 e) {
  v16h r;
#pragma unroll
  for (int i = 0; i < 16; ++i) r[i] = e;
  return r;
}

// ---------------- WMMA fragments (16x16x32 f16, wave32) --------------------
// A (16xK=32): lanes 0-15 -> M=lane, elems 0-15 = K 0..15;
// lanes 16-31 -> K 16..31 (contiguous, matches B fragments).
__device__ __forceinline__ v16h load_a_frag(const _Float16* W, int lda,
                                            int m0, int kb, int lane) {
  int m  = m0 + (lane & 15);
  int kh = (lane >> 4) & 1;
  const _Float16* r = W + m * lda + kb + kh * 16;
  v8h lo = *(const v8h*)r;
  v8h hi = *(const v8h*)(r + 8);
  return __builtin_shufflevector(lo, hi, 0,1,2,3,4,5,6,7,8,9,10,11,12,13,14,15);
}

// Preload all W fragments into registers: A[kt][mt], kt = K/32 step, mt = M/16.
template <int NKT>
__device__ __forceinline__ void load_afrags(const _Float16* W, int lda, int lane,
                                            v16h (&A)[NKT][4]) {
#pragma unroll
  for (int kt = 0; kt < NKT; ++kt)
#pragma unroll
    for (int mt = 0; mt < 4; ++mt)
      A[kt][mt] = load_a_frag(W, lda, mt * 16, kt * 32, lane);
}

// One 16-column tile of Out = relu(W[64 x NKT*32] @ B).  B is fetched by a
// functor bfrag(half, cseg, n, node) returning 16 channel-contiguous halves:
//   half = which 64-row operand (0: rows 0-63, 1: rows 64-127)
//   cseg = 16-channel segment within the operand (0..3)
// Stores via store(n, mbase, v8f): 8 contiguous output channels per call.
template <int NKT, typename FB, typename FS>
__device__ __forceinline__ void gemm_tile(const v16h (&A)[NKT][4],
                                          FB bfrag, FS store, int col0, int lane) {
  const v8f z8 = {0.f,0.f,0.f,0.f,0.f,0.f,0.f,0.f};
  v8f acc[4] = {z8, z8, z8, z8};
  int kh   = (lane >> 4) & 1;
  int n    = col0 + (lane & 15);
  int node = n / KN;                       // hoisted: used only by op3
#pragma unroll
  for (int kt = 0; kt < NKT; ++kt) {
    int half = kt >> 1;                    // kb = kt*32
    int cseg = ((kt & 1) << 1) + kh;       // ((kb&63)>>4) + kh
    v16h b = bfrag(half, cseg, n, node);
#pragma unroll
    for (int mt = 0; mt < 4; ++mt)
      acc[mt] = __builtin_amdgcn_wmma_f32_16x16x32_f16(
          false, A[kt][mt], false, b, (short)0, acc[mt], false, false);
  }
#pragma unroll
  for (int mt = 0; mt < 4; ++mt) {
    v8f r;
#pragma unroll
    for (int e = 0; e < 8; ++e) r[e] = fmaxf(acc[mt][e], 0.0f);
    store(n, mt * 16 + 8 * kh, r);         // 8 contiguous channels
  }
}

// atom-op fragment: 16 channels [cseg*16 .. +15] of op at column n.
// Layouts are channel-contiguous: Xi/Xj[col][64], Mi[node][64] f16, Eud[col].
__device__ __forceinline__ v16h op_frag(int op, int n, int node, int cseg,
                                        const _Float16* Xi, const _Float16* Xj,
                                        const _Float16* Mi, const _Float16* Eud) {
  const _Float16* xi = Xi + n * CH + cseg * 16;
  switch (op) {
    case 0:  return ld16(xi);
    case 1:  return ld16(Xj + n * CH + cseg * 16);
    case 2:  return ld16(xi) - ld16(Xj + n * CH + cseg * 16);
    case 3:  return ld16(xi) - ld16(Mi + node * CH + cseg * 16);
    default: return splat16(Eud[n]);       // constant over channels
  }
}

// ---------------- gather + derived-op prep ---------------------------------
__device__ __forceinline__ void gather_xy(int b, int nodeBase,
                                          const float* x, const long long* ei,
                                          int* nbr0, int* nbr1,
                                          _Float16* Xi, _Float16* Xj, int tid) {
  for (int q = tid; q < 2 * COLS; q += NTHREADS) {
    int e = q / COLS, col = q % COLS;
    int node = col / KN, k = col % KN;
    int n = nodeBase + node;
    long long v = (k < KE)
        ? ei[(((long long)e * NBATCH + b) * NNODES + n) * KE + k]
        : (long long)n;                       // self loop at k == 16
    (e ? nbr1 : nbr0)[col] = (int)v;
  }
  __syncthreads();
  for (int it = tid; it < CH * COLS; it += NTHREADS) {
    int col = it >> 6, c = it & 63;           // Xi[col][c]
    const float* xb = x + ((size_t)(b * CH + c)) * NNODES;   // x[b,c,:,0]
    Xi[it] = (_Float16)xb[nbr1[col]];         // ei[1] -> x_i
    Xj[it] = (_Float16)xb[nbr0[col]];         // ei[0] -> x_j
  }
}

__device__ __forceinline__ void prep_stats(const _Float16* Xi, const _Float16* Xj,
                                           _Float16* Mi, _Float16* Eud, int tid) {
  for (int it = tid; it < CH * NB; it += NTHREADS) {   // mean over K -> Mi[node][c]
    int node = it >> 6, c = it & 63;
    float s = 0.f;
#pragma unroll
    for (int k = 0; k < KN; ++k) s += (float)Xi[(node * KN + k) * CH + c];
    Mi[it] = (_Float16)(s * (1.0f / KN));
  }
  for (int col = tid; col < COLS; col += NTHREADS) {   // euclid over C
    const _Float16* xi = Xi + col * CH;
    const _Float16* xj = Xj + col * CH;
    float s = 0.f;
    for (int c = 0; c < CH; ++c) {
      float d = (float)xi[c] - (float)xj[c];
      s += d * d;
    }
    Eud[col] = (_Float16)sqrtf(s);
  }
}

// ---------------- kernel 0: zero score accumulators ------------------------
__global__ void k_init(float* S) {
  if (threadIdx.x < 10) S[threadIdx.x] = 0.f;
}

// ---------------- kernel 1: alpha logits A = sum_b relu(W@[xi;xj]) ---------
__global__ __launch_bounds__(NTHREADS, 1)
void k1_alpha_logits(const float* __restrict__ x, const long long* __restrict__ ei,
                     const float* __restrict__ Wnn, float* __restrict__ A_ws) {
  extern __shared__ char smem[];
  char* p = smem;
  int*      nbr0 = (int*)p;      p += COLS * 4;
  int*      nbr1 = (int*)p;      p += COLS * 4;
  _Float16* Wh   = (_Float16*)p; p += CH * CH2 * 2;
  _Float16* Xi   = (_Float16*)p; p += COLS * CH * 2;
  _Float16* Xj   = (_Float16*)p; p += COLS * CH * 2;
  float*    Ab   = (float*)p;    p += COLS * CH * 4;   // Ab[col][c] f32

  int tid = threadIdx.x, lane = tid & 31, wave = tid >> 5;
  int nodeBase = blockIdx.x * NB;

  for (int it = tid; it < CH * CH2; it += NTHREADS) Wh[it] = (_Float16)Wnn[it];
  for (int it = tid; it < COLS * CH; it += NTHREADS) Ab[it] = 0.f;
  __syncthreads();

  v16h AF[4][4];
  load_afrags<4>(Wh, CH2, lane, AF);      // W_nn fragments register-resident

  for (int b = 0; b < NBATCH; ++b) {
    gather_xy(b, nodeBase, x, ei, nbr0, nbr1, Xi, Xj, tid);
    __syncthreads();
    auto bv = [&](int half, int cseg, int n, int node) -> v16h {
      return ld16((half ? Xj : Xi) + n * CH + cseg * 16);
    };
    auto st = [&](int n, int mbase, v8f v) {
      v8f* q = (v8f*)(Ab + n * CH + mbase);
      *q += v;
    };
    for (int tile = wave; tile < NTILE; tile += 8)
      gemm_tile<4>(AF, bv, st, tile * 16, lane);
    __syncthreads();
  }
  for (int it = tid; it < COLS * CH; it += NTHREADS) {
    int col = it >> 6, c = it & 63;
    int node = col / KN, k = col % KN;
    A_ws[((size_t)c * NNODES + nodeBase + node) * KN + k] = Ab[it];
  }
}

// ---------------- kernel 2: fused pair/triple convs + score reduction ------
__global__ __launch_bounds__(NTHREADS, 1)
void k2_scores(const float* __restrict__ x, const long long* __restrict__ ei,
               const float* __restrict__ Wnn, const float* __restrict__ att,
               const float* __restrict__ A_ws, float* __restrict__ S) {
  extern __shared__ char smem[];
  char* p = smem;
  int*      nbr0 = (int*)p;      p += COLS * 4;
  int*      nbr1 = (int*)p;      p += COLS * 4;
  _Float16* Wh   = (_Float16*)p; p += CH * CH2 * 2;
  _Float16* Xi   = (_Float16*)p; p += COLS * CH * 2;
  _Float16* Xj   = (_Float16*)p; p += COLS * CH * 2;
  _Float16* Mi   = (_Float16*)p; p += NB * CH * 2;
  _Float16* Eud  = (_Float16*)p; p += COLS * 2;
  _Float16* Pb   = (_Float16*)p; p += COLS * CH * 2;   // pair conv   Pb[col][c]
  _Float16* Tb   = (_Float16*)p; p += COLS * CH * 2;   // triple conv Tb[col][c]
  float*    sl   = (float*)p;    p += 16 * 4;

  int tid = threadIdx.x, lane = tid & 31, wave = tid >> 5;
  int nodeBase = blockIdx.x * NB;
  if (tid < 10) sl[tid] = 0.f;
  for (int it = tid; it < CH * CH2; it += NTHREADS) Wh[it] = (_Float16)Wnn[it];
  __syncthreads();

  v16h AF[4][4];
  load_afrags<4>(Wh, CH2, lane, AF);      // shared by all 16 convs

  const int PI[6] = {0,0,0,1,1,2};
  const int PJ[6] = {1,2,3,2,3,3};

  auto stP = [&](int n, int mbase, v8f v) {
    v8h h;
#pragma unroll
    for (int e = 0; e < 8; ++e) h[e] = (_Float16)v[e];
    *(v8h*)(Pb + n * CH + mbase) = h;      // ds_store_b128
  };
  auto stT = [&](int n, int mbase, v8f v) {
    v8h h;
#pragma unroll
    for (int e = 0; e < 8; ++e) h[e] = (_Float16)v[e];
    *(v8h*)(Tb + n * CH + mbase) = h;
  };

  for (int b = 0; b < NBATCH; ++b) {
    __syncthreads();
    gather_xy(b, nodeBase, x, ei, nbr0, nbr1, Xi, Xj, tid);
    __syncthreads();
    prep_stats(Xi, Xj, Mi, Eud, tid);
    __syncthreads();

    int t = 0;
    for (int pp = 0; pp < 6; ++pp) {
      int pi = PI[pp], pj = PJ[pp];
      auto bvP = [&](int half, int cseg, int n, int node) -> v16h {
        return op_frag(half ? pj : pi, n, node, cseg, Xi, Xj, Mi, Eud);
      };
      for (int tile = wave; tile < NTILE; tile += 8)
        gemm_tile<4>(AF, bvP, stP, tile * 16, lane);
      __syncthreads();

      for (int kk = pj + 1; kk < 5; ++kk, ++t) {
        auto bvT = [&](int half, int cseg, int n, int node) -> v16h {
          return half ? op_frag(kk, n, node, cseg, Xi, Xj, Mi, Eud)
                      : ld16(Pb + n * CH + cseg * 16);
        };
        for (int tile = wave; tile < NTILE; tile += 8)
          gemm_tile<4>(AF, bvT, stT, tile * 16, lane);
        __syncthreads();

        // s_t += sum_{c,node} softmax_k(lrelu(att_t*A)) . T  (scale-free argmax)
        float at = att[t];
        float local = 0.f;
        for (int it = tid; it < CH * NB; it += NTHREADS) {
          int c = it >> 4, node = it & 15;
          const float* Ar = A_ws + ((size_t)c * NNODES + nodeBase + node) * KN;
          float z[KN];
          float mx = -3.4e38f;
#pragma unroll
          for (int k = 0; k < KN; ++k) {
            float v = at * Ar[k];
            v = (v > 0.f) ? v : 0.2f * v;      // leaky relu 0.2
            z[k] = v;
            mx = fmaxf(mx, v);
          }
          float den = 0.f, num = 0.f;
          const _Float16* Tr = Tb + node * KN * CH + c;
#pragma unroll
          for (int k = 0; k < KN; ++k) {
            float e = __expf(z[k] - mx);
            den += e;
            num += e * (float)Tr[k * CH];
          }
          local += num / den;
        }
        atomicAdd(&sl[t], local);
        __syncthreads();
      }
    }
  }
  __syncthreads();
  if (tid < 10) atomicAdd(&S[tid], sl[tid]);
}

// ---------------- kernel 3: argmax over 10 scores --------------------------
__global__ void k3_argmax(const float* S, int* idxp) {
  if (threadIdx.x == 0) {
    int best = 0;
    float bv = S[0];
    for (int t = 1; t < 10; ++t)
      if (S[t] > bv) { bv = S[t]; best = t; }
    *idxp = best;
  }
}

// ---------------- kernel 4: recompute winner, W_out conv, max over K -------
__global__ __launch_bounds__(NTHREADS, 1)
void k4_output(const float* __restrict__ x, const long long* __restrict__ ei,
               const float* __restrict__ Wnn, const float* __restrict__ Wout,
               const int* __restrict__ selp, float* __restrict__ out) {
  extern __shared__ char smem[];
  char* p = smem;
  int*      nbr0 = (int*)p;      p += COLS * 4;
  int*      nbr1 = (int*)p;      p += COLS * 4;
  _Float16* Wh   = (_Float16*)p; p += CH * CH2 * 2;
  _Float16* Wo   = (_Float16*)p; p += CH * CH * 2;
  _Float16* Xi   = (_Float16*)p; p += COLS * CH * 2;
  _Float16* Xj   = (_Float16*)p; p += COLS * CH * 2;
  _Float16* Mi   = (_Float16*)p; p += NB * CH * 2;
  _Float16* Eud  = (_Float16*)p; p += COLS * 2;
  _Float16* Pb   = (_Float16*)p; p += COLS * CH * 2;
  _Float16* Tb   = (_Float16*)p; p += COLS * CH * 2;

  int tid = threadIdx.x, lane = tid & 31, wave = tid >> 5;
  int nodeBase = blockIdx.x * NB;
  int b = blockIdx.y;

  int tsel = *selp;                        // uniform device-side selection
  int oi = c_TI[tsel], oj = c_TJ[tsel], ok = c_TK[tsel];

  for (int it = tid; it < CH * CH2; it += NTHREADS) Wh[it] = (_Float16)Wnn[it];
  for (int it = tid; it < CH * CH;  it += NTHREADS) Wo[it] = (_Float16)Wout[it];
  __syncthreads();

  v16h AF[4][4];
  load_afrags<4>(Wh, CH2, lane, AF);

  gather_xy(b, nodeBase, x, ei, nbr0, nbr1, Xi, Xj, tid);
  __syncthreads();
  prep_stats(Xi, Xj, Mi, Eud, tid);
  __syncthreads();

  auto bvP = [&](int half, int cseg, int n, int node) -> v16h {
    return op_frag(half ? oj : oi, n, node, cseg, Xi, Xj, Mi, Eud);
  };
  auto stP = [&](int n, int mbase, v8f v) {
    v8h h;
#pragma unroll
    for (int e = 0; e < 8; ++e) h[e] = (_Float16)v[e];
    *(v8h*)(Pb + n * CH + mbase) = h;
  };
  for (int tile = wave; tile < NTILE; tile += 8)
    gemm_tile<4>(AF, bvP, stP, tile * 16, lane);
  __syncthreads();

  auto bvT = [&](int half, int cseg, int n, int node) -> v16h {
    return half ? op_frag(ok, n, node, cseg, Xi, Xj, Mi, Eud)
                : ld16(Pb + n * CH + cseg * 16);
  };
  auto stT = [&](int n, int mbase, v8f v) {
    v8h h;
#pragma unroll
    for (int e = 0; e < 8; ++e) h[e] = (_Float16)v[e];
    *(v8h*)(Tb + n * CH + mbase) = h;
  };
  for (int tile = wave; tile < NTILE; tile += 8)
    gemm_tile<4>(AF, bvT, stT, tile * 16, lane);
  __syncthreads();

  // final conv: Out = relu(W_out[64x64] @ T); Ob aliases Xi (dead now)
  _Float16* Ob = Xi;
  v16h AO[2][4];
  load_afrags<2>(Wo, CH, lane, AO);
  auto bvO = [&](int half, int cseg, int n, int node) -> v16h {
    (void)half; (void)node;
    return ld16(Tb + n * CH + cseg * 16);
  };
  auto stO = [&](int n, int mbase, v8f v) {
    v8h h;
#pragma unroll
    for (int e = 0; e < 8; ++e) h[e] = (_Float16)v[e];
    *(v8h*)(Ob + n * CH + mbase) = h;
  };
  for (int tile = wave; tile < NTILE; tile += 8)
    gemm_tile<2>(AO, bvO, stO, tile * 16, lane);
  __syncthreads();

  // max over K (relu output nonnegative)
  for (int it = tid; it < CH * NB; it += NTHREADS) {
    int c = it >> 4, node = it & 15;
    const _Float16* Or = Ob + node * KN * CH + c;
    float m = 0.f;
#pragma unroll
    for (int k = 0; k < KN; ++k) m = fmaxf(m, (float)Or[k * CH]);
    out[((size_t)(b * CH + c)) * NNODES + nodeBase + node] = m;
  }
}

// ---------------- host launcher --------------------------------------------
extern "C" void kernel_launch(void* const* d_in, const int* in_sizes, int n_in,
                              void* d_out, int out_size, void* d_ws, size_t ws_size,
                              hipStream_t stream) {
  (void)in_sizes; (void)n_in; (void)out_size; (void)ws_size;
  const float*     x    = (const float*)d_in[0];       // [2,64,4096,1] f32
  const long long* ei   = (const long long*)d_in[1];   // [2,2,4096,16] i64
  const float*     Wnn  = (const float*)d_in[2];       // [64,128] f32
  const float*     Wout = (const float*)d_in[3];       // [64,64]  f32
  const float*     att  = (const float*)d_in[4];       // [10] f32
  float*           out  = (float*)d_out;               // [2,64,4096,1] f32

  char* ws = (char*)d_ws;
  float* A_ws = (float*)ws;                                        // 17.8 MB
  size_t A_bytes = (size_t)CH * NNODES * KN * sizeof(float);
  float* S    = (float*)(ws + A_bytes);                            // 10 f32
  int*   idxp = (int*)(ws + A_bytes + 64);

  size_t nbr   = 2 * (COLS * 4);
  size_t wh    = (size_t)CH * CH2 * 2;
  size_t xmat  = (size_t)COLS * CH * 2;          // one f16 [col][64] matrix
  size_t mi    = (size_t)NB * CH * 2;
  size_t eud   = (size_t)COLS * 2;

  size_t smem1 = nbr + wh + 2 * xmat + (size_t)COLS * CH * 4;            // +Ab f32
  size_t smem2 = nbr + wh + 2 * xmat + mi + eud + 2 * xmat + 64;         // +Pb,Tb,sl
  size_t smem4 = nbr + wh + (size_t)CH * CH * 2 + 2 * xmat + mi + eud
               + 2 * xmat;                                               // +Wo,Pb,Tb

  int nblocks = NNODES / NB;   // 256

  k_init<<<1, 32, 0, stream>>>(S);
  k1_alpha_logits<<<nblocks, NTHREADS, smem1, stream>>>(x, ei, Wnn, A_ws);
  k2_scores<<<nblocks, NTHREADS, smem2, stream>>>(x, ei, Wnn, att, A_ws, S);
  k3_argmax<<<1, 32, 0, stream>>>(S, idxp);
  k4_output<<<dim3(nblocks, NBATCH), NTHREADS, smem4, stream>>>(x, ei, Wnn, Wout,
                                                                idxp, out);
}
